// Tokenizer_53283364274528
// MI455X (gfx1250) — matrix operand order repaired and checked by reading, not compiled, not run on hardware
//
#include <hip/hip_runtime.h>

// ---------------------------------------------------------------------------
// CDNA5 (gfx1250) WMMA types
// ---------------------------------------------------------------------------
typedef __attribute__((ext_vector_type(16))) __bf16 v16bf;
typedef __attribute__((ext_vector_type(8)))  float  v8f;

union Frag {
    v16bf v;
    uint4 q[2];
    unsigned short u[16];
};

__device__ __forceinline__ unsigned short f2bf(float f) {
    unsigned int x = __float_as_uint(f);
    x += 0x7FFFu + ((x >> 16) & 1u);          // round-to-nearest-even
    return (unsigned short)(x >> 16);
}

// ---------------------------------------------------------------------------
// Embedding gather: ids (B,62,62,4) -> padded NHWC bf16 grid (B,66,66,128)
// ---------------------------------------------------------------------------
__global__ void k_embed(const int* __restrict__ x, const float* __restrict__ emb,
                        unsigned short* __restrict__ g, long total) {
    for (long i = (long)blockIdx.x * blockDim.x + threadIdx.x; i < total;
         i += (long)gridDim.x * blockDim.x) {
        int c = (int)(i & 127);
        long rest = i >> 7;
        int w = (int)(rest % 62); rest /= 62;
        int h = (int)(rest % 62);
        int img = (int)(rest / 62);
        int s = c >> 5, e = c & 31;
        int id = x[(((long)img * 62 + h) * 62 + w) * 4 + s];
        float v = emb[id * 32 + e];
        g[(((long)img * 66 + (h + 2)) * 66 + (w + 2)) * 128 + c] = f2bf(v);
    }
}

// ---------------------------------------------------------------------------
// Weight shuffle: OIHW f32 -> per-lane WMMA B-fragment bf16 layout
// frag[t][kc][nt][lane][j], K = kc*32 + (lane>=16 ? 16 : 0) + j,
// N = nt*16 + (lane&15).  (Contiguous k order, matching the A-fragment loads.)
// ---------------------------------------------------------------------------
__global__ void k_prep(const float* __restrict__ W, unsigned short* __restrict__ frag,
                       int ks, int Cin, int Cout, int KC, int NT, long total) {
    for (long i = (long)blockIdx.x * blockDim.x + threadIdx.x; i < total;
         i += (long)gridDim.x * blockDim.x) {
        int j = (int)(i & 15);
        int lane = (int)((i >> 4) & 31);
        long r = i >> 9;
        int nt = (int)(r % NT); r /= NT;
        int kc = (int)(r % KC); r /= KC;
        int t = (int)r;
        int half = lane >> 4;
        int n = nt * 16 + (lane & 15);
        int k = kc * 32 + half * 16 + j;
        int ky = t / ks, kx = t % ks;
        float v = 0.f;
        if (n < Cout && k < Cin)
            v = W[(((long)n * Cin + k) * ks + ky) * ks + kx];
        frag[i] = f2bf(v);
    }
}

// ---------------------------------------------------------------------------
// Codebook squared norms ||c_n||^2
// ---------------------------------------------------------------------------
__global__ void k_c2(const float* __restrict__ cb, float* __restrict__ c2) {
    int n = blockIdx.x * blockDim.x + threadIdx.x;
    if (n < 1024) {
        float s = 0.f;
        for (int d = 0; d < 256; ++d) { float v = cb[n * 256 + d]; s += v * v; }
        c2[n] = s;
    }
}

// ---------------------------------------------------------------------------
// Implicit-GEMM conv via v_wmma_f32_16x16x32_bf16.
// Block = 4 waves (128 thr): block tile = 64 output pixels x 16 channels.
// The per-nt weight-fragment slab (taps*KC KB) is staged global->LDS once per
// block with CDNA5 async-to-LDS copies (ASYNCcnt), then B fragments are served
// by ds_load_b128, amortized over 4 M-tiles.  A fragments stream from the
// padded NHWC bf16 activation buffer.  ostride=2 writes zero-stuffed buffers
// that turn the transposed convs into ordinary stride-1 convs.
// ---------------------------------------------------------------------------
__global__ void k_conv(const unsigned short* __restrict__ in,
                       const unsigned short* __restrict__ wf,
                       const float* __restrict__ bias,
                       unsigned short* __restrict__ out,
                       float* __restrict__ aux,
                       int Hp, int Wp, int Cin,
                       int Ho, int Wo, int Cout,
                       int stride, int ks,
                       int Wop, long outImgStride,
                       int padT, int padL, int ostride,
                       int relu, int auxLayout)
{
    extern __shared__ unsigned short smem[];   // [taps*KC][32 lanes][16 bf16]

    const int tid  = threadIdx.x;
    const int lane = tid & 31;
    const int wid  = tid >> 5;
    const int row  = lane & 15;
    const int half = lane >> 4;
    const int img  = blockIdx.z;
    const int nt   = blockIdx.y;
    const int NT   = gridDim.y;
    const int KC   = Cin >> 5;
    const int HoWo = Ho * Wo;

    // ---- stage this nt's weight fragments into LDS (async, ASYNCcnt) ----
    const int chunks = ks * ks * KC * 64;      // 16B chunks
    for (int i = tid; i < chunks; i += 128) {
        int tkc = i >> 6, within = i & 63;
        const unsigned short* gp = wf + ((long)tkc * NT + nt) * 512 + within * 8;
        unsigned lds_off = (unsigned)(unsigned long long)(const void*)(smem + i * 8);
        asm volatile("global_load_async_to_lds_b128 %0, %1, off"
                     :: "v"(lds_off), "v"(gp) : "memory");
    }
    asm volatile("s_wait_asynccnt 0x0" ::: "memory");
    __syncthreads();

    // ---- per-wave 16x16 output tile -------------------------------------
    const int m0 = (blockIdx.x * 4 + wid) * 16;
    int m = m0 + row;
    if (m >= HoWo) m = HoWo - 1;               // clamp for safe A loads
    const int oh = m / Wo, ow = m % Wo;
    const unsigned short* inImg = in + (long)img * Hp * Wp * Cin;
    const int ih0 = oh * stride, iw0 = ow * stride;

    v8f acc = {};
    Frag a, b;
    for (int ky = 0; ky < ks; ++ky) {
        for (int kx = 0; kx < ks; ++kx) {
            const int t = ky * ks + kx;
            const unsigned short* ap0 =
                inImg + ((long)(ih0 + ky) * Wp + (iw0 + kx)) * Cin + half * 16;
            for (int kc = 0; kc < KC; ++kc) {
                const unsigned short* ap = ap0 + kc * 32;
                a.q[0] = *(const uint4*)(ap);
                a.q[1] = *(const uint4*)(ap + 8);
                const unsigned short* bp = smem + ((t * KC + kc) * 512 + lane * 16);
                b.q[0] = *(const uint4*)(bp);      // ds_load_b128
                b.q[1] = *(const uint4*)(bp + 8);
                acc = __builtin_amdgcn_wmma_f32_16x16x32_bf16(
                    false, a.v, false, b.v, (short)0, acc, false, false);
            }
        }
    }

    const int ch = nt * 16 + row;
    if (ch >= Cout) return;
    const float bv = bias[ch];
    for (int r = 0; r < 8; ++r) {
        int mm = m0 + r + half * 8;            // C/D layout: M = r + 8*half
        if (mm >= HoWo) continue;
        float v = acc[r] + bv;
        if (relu && v < 0.f) v = 0.f;
        int oh2 = mm / Wo, ow2 = mm % Wo;
        if (out)
            out[(long)img * outImgStride +
                ((long)(padT + oh2 * ostride) * Wop + (padL + ow2 * ostride)) * Cout + ch] = f2bf(v);
        if (aux) {
            long ai = (auxLayout == 0)
                ? (((long)img * Cout + ch) * Ho + oh2) * Wo + ow2     // NCHW (z)
                : (((long)img * Ho + oh2) * Wo + ow2) * Cout + ch;    // NHWC (logits)
            aux[ai] = v;
        }
    }
}

// ---------------------------------------------------------------------------
// VQ: score = ||c_n||^2 - 2*z.c_n, WMMA GEMM per 16-code tile + running
// argmin, final 16-lane shuffle-xor reduction.  Codebook slab (512KB) exceeds
// LDS and is shared by all 2048 blocks -> stream it from L2.
// ---------------------------------------------------------------------------
__global__ void k_vq(const unsigned short* __restrict__ z,
                     const unsigned short* __restrict__ cbf,
                     const float* __restrict__ c2,
                     int* __restrict__ tok, float* __restrict__ tokF)
{
    const int lane = threadIdx.x;
    const int row  = lane & 15;
    const int half = lane >> 4;
    const long r0  = (long)blockIdx.x * 16;

    const unsigned short* zp = z + (r0 + row) * 256 + half * 16;
    Frag a[8];
    for (int kc = 0; kc < 8; ++kc) {
        const unsigned short* ap = zp + kc * 32;
        a[kc].q[0] = *(const uint4*)(ap);
        a[kc].q[1] = *(const uint4*)(ap + 8);
    }

    float minv[8]; int mini[8];
    for (int r = 0; r < 8; ++r) { minv[r] = 3.4e38f; mini[r] = 0; }

    for (int ntile = 0; ntile < 64; ++ntile) {
        v8f acc = {};
        Frag b;
        for (int kc = 0; kc < 8; ++kc) {
            const unsigned short* bp = cbf + (((long)kc * 64 + ntile) * 32 + lane) * 16;
            b.q[0] = *(const uint4*)(bp);
            b.q[1] = *(const uint4*)(bp + 8);
            acc = __builtin_amdgcn_wmma_f32_16x16x32_bf16(
                false, a[kc].v, false, b.v, (short)0, acc, false, false);
        }
        const int n = ntile * 16 + row;
        const float cc = c2[n];
        for (int r = 0; r < 8; ++r) {
            float s = cc - 2.f * acc[r];
            if (s < minv[r]) { minv[r] = s; mini[r] = n; }   // strict '<' keeps first-min
        }
    }

    for (int r = 0; r < 8; ++r) {
        float v = minv[r]; int i = mini[r];
        for (int off = 1; off < 16; off <<= 1) {             // within half-wave group
            float ov = __shfl_xor(v, off, 32);
            int   oi = __shfl_xor(i, off, 32);
            if (ov < v || (ov == v && oi < i)) { v = ov; i = oi; }
        }
        if (row == 0) {
            long rr = r0 + r + half * 8;
            tok[rr]  = i;
            tokF[rr] = (float)i;
        }
    }
}

// ---------------------------------------------------------------------------
// Gather: z_q = codebook[tokens]; f32 NCHW to d_out + bf16 rows for decoder.
// ---------------------------------------------------------------------------
__global__ void k_gather(const int* __restrict__ tok, const float* __restrict__ cb,
                         float* __restrict__ zq, unsigned short* __restrict__ zqb,
                         long total)
{
    for (long idx = (long)blockIdx.x * blockDim.x + threadIdx.x; idx < total;
         idx += (long)gridDim.x * blockDim.x) {
        int d = (int)(idx & 255);
        long row = idx >> 8;
        int t = tok[row];
        float v = cb[(long)t * 256 + d];
        zqb[idx] = f2bf(v);
        int img = (int)(row >> 8);
        int p = (int)(row & 255);
        int h = p >> 4, w = p & 15;
        zq[(((long)img * 256 + d) * 16 + h) * 16 + w] = v;
    }
}

// ---------------------------------------------------------------------------
// Host-side pipeline
// ---------------------------------------------------------------------------
extern "C" void kernel_launch(void* const* d_in, const int* in_sizes, int n_in,
                              void* d_out, int out_size, void* d_ws, size_t ws_size,
                              hipStream_t stream)
{
    (void)in_sizes; (void)n_in; (void)out_size; (void)ws_size;

    const int*   x        = (const int*)  d_in[0];
    const float* id_emb   = (const float*)d_in[1];
    const float* enc_w1   = (const float*)d_in[2];
    const float* enc_b1   = (const float*)d_in[3];
    const float* enc_w2   = (const float*)d_in[4];
    const float* enc_b2   = (const float*)d_in[5];
    const float* enc_w3   = (const float*)d_in[6];
    const float* enc_b3   = (const float*)d_in[7];
    const float* preq_w   = (const float*)d_in[8];
    const float* preq_b   = (const float*)d_in[9];
    const float* codebook = (const float*)d_in[10];
    const float* postq_w  = (const float*)d_in[11];
    const float* postq_b  = (const float*)d_in[12];
    const float* dec_w1   = (const float*)d_in[13];
    const float* dec_b1   = (const float*)d_in[14];
    const float* dec_w2   = (const float*)d_in[15];
    const float* dec_b2   = (const float*)d_in[16];
    const float* dec_w3   = (const float*)d_in[17];
    const float* dec_b3   = (const float*)d_in[18];
    const float* out_w    = (const float*)d_in[19];
    const float* out_b    = (const float*)d_in[20];

    float* outF    = (float*)d_out;
    float* z_out   = outF;                         // (128,256,16,16)
    float* zq_out  = outF + 8388608;               // (128,256,16,16)
    float* tokF    = outF + 16777216;              // (128,256)
    float* logits  = outF + 16809984;              // (128,62,62,68)

    char* ws = (char*)d_ws;
    size_t off = 0;
    auto alloc = [&](size_t bytes) {
        size_t o = off; off += (bytes + 255) & ~(size_t)255; return o;
    };

    // Activation arenas (bf16), reused across pipeline phases.
    const size_t A0_B = (size_t)66 * 66 * 128 * 128 * 2;   // g / S0(stuffed) / D2
    const size_t A1_B = (size_t)66 * 66 *  64 * 128 * 2;   // enc1-out / S1(stuffed) / D3
    const size_t A2_B = (size_t)34 * 34 *  64 * 128 * 2;   // enc2-out
    const size_t A3_B = (size_t)16 * 16 * 256 * 128 * 2;   // enc3-out
    unsigned short* A0 = (unsigned short*)(ws + alloc(A0_B));
    unsigned short* A1 = (unsigned short*)(ws + alloc(A1_B));
    unsigned short* A2 = (unsigned short*)(ws + alloc(A2_B));
    unsigned short* A3 = (unsigned short*)(ws + alloc(A3_B));
    unsigned short* Zb = (unsigned short*)(ws + alloc(A3_B));  // z bf16 (VQ input)
    unsigned short* ZQ = (unsigned short*)(ws + alloc(A3_B));  // z_q bf16 (decoder input)

    auto frB = [](int taps, int KC, int NT) { return (size_t)taps * KC * NT * 512 * 2; };
    unsigned short* F_e1 = (unsigned short*)(ws + alloc(frB(9, 4,  4)));
    unsigned short* F_e2 = (unsigned short*)(ws + alloc(frB(9, 2,  4)));
    unsigned short* F_e3 = (unsigned short*)(ws + alloc(frB(9, 2, 16)));
    unsigned short* F_pq = (unsigned short*)(ws + alloc(frB(1, 8, 16)));
    unsigned short* F_cb = (unsigned short*)(ws + alloc(frB(1, 8, 64)));
    unsigned short* F_qp = (unsigned short*)(ws + alloc(frB(1, 8, 16)));
    unsigned short* F_d1 = (unsigned short*)(ws + alloc(frB(9, 8,  4)));
    unsigned short* F_d2 = (unsigned short*)(ws + alloc(frB(9, 2,  4)));
    unsigned short* F_d3 = (unsigned short*)(ws + alloc(frB(9, 2,  4)));
    unsigned short* F_ow = (unsigned short*)(ws + alloc(frB(1, 2,  5)));

    int*   tokI = (int*)  (ws + alloc(32768 * 4));
    float* c2   = (float*)(ws + alloc(1024 * 4));

    auto prep = [&](const float* W, unsigned short* frag, int ks, int Cin, int Cout,
                    int KC, int NT) {
        long total = (long)ks * ks * KC * NT * 512;
        int blocks = (int)((total + 255) / 256);
        k_prep<<<blocks, 256, 0, stream>>>(W, frag, ks, Cin, Cout, KC, NT, total);
    };
    auto conv = [&](const unsigned short* in, const unsigned short* wf, const float* bias,
                    unsigned short* outb, float* aux,
                    int Hp, int Wp, int Cin, int Ho, int Wo, int Cout, int stride, int ks,
                    int Wop, long outImgStride, int padT, int padL, int ostride,
                    int relu, int auxLayout) {
        dim3 g((Ho * Wo + 63) / 64, (Cout + 15) / 16, 128);
        size_t lds = (size_t)ks * ks * (Cin >> 5) * 1024;   // B-frag slab per nt
        k_conv<<<g, 128, lds, stream>>>(in, wf, bias, outb, aux, Hp, Wp, Cin, Ho, Wo, Cout,
                                        stride, ks, Wop, outImgStride, padT, padL, ostride,
                                        relu, auxLayout);
    };

    // --- weight shuffles + codebook norms -------------------------------
    prep(enc_w1,   F_e1, 3, 128,   64, 4,  4);
    prep(enc_w2,   F_e2, 3,  64,   64, 2,  4);
    prep(enc_w3,   F_e3, 3,  64,  256, 2, 16);
    prep(preq_w,   F_pq, 1, 256,  256, 8, 16);
    prep(codebook, F_cb, 1, 256, 1024, 8, 64);
    prep(postq_w,  F_qp, 1, 256,  256, 8, 16);
    prep(dec_w1,   F_d1, 3, 256,   64, 8,  4);
    prep(dec_w2,   F_d2, 3,  64,   64, 2,  4);
    prep(dec_w3,   F_d3, 3,  64,   64, 2,  4);
    prep(out_w,    F_ow, 1,  64,   68, 2,  5);
    k_c2<<<4, 256, 0, stream>>>(codebook, c2);

    // --- encoder --------------------------------------------------------
    hipMemsetAsync(A0, 0, A0_B, stream);                 // halo (pad=2)
    {
        long total = (long)128 * 62 * 62 * 128;
        k_embed<<<(int)((total + 255) / 256), 256, 0, stream>>>(x, id_emb, A0, total);
    }
    hipMemsetAsync(A1, 0, A1_B, stream);                 // halo (pad=1)
    conv(A0, F_e1, enc_b1, A1, nullptr, 66, 66, 128, 64, 64, 64, 1, 3,
         66, (long)66 * 66 * 64, 1, 1, 1, 1, 0);         // 62->64
    hipMemsetAsync(A2, 0, A2_B, stream);                 // halo (pad=1)
    conv(A1, F_e2, enc_b2, A2, nullptr, 66, 66, 64, 32, 32, 64, 2, 3,
         34, (long)34 * 34 * 64, 1, 1, 1, 1, 0);         // 64->32
    conv(A2, F_e3, enc_b3, A3, nullptr, 34, 34, 64, 16, 16, 256, 2, 3,
         16, (long)16 * 16 * 256, 0, 0, 1, 1, 0);        // 32->16
    conv(A3, F_pq, preq_b, Zb, z_out, 16, 16, 256, 16, 16, 256, 1, 1,
         16, (long)16 * 16 * 256, 0, 0, 1, 0, 0);        // z (+ f32 NCHW out)

    // --- VQ -------------------------------------------------------------
    k_vq<<<2048, 32, 0, stream>>>(Zb, F_cb, c2, tokI, tokF);
    {
        long total = (long)32768 * 256;
        k_gather<<<(int)((total + 255) / 256), 256, 0, stream>>>(tokI, codebook,
                                                                 zq_out, ZQ, total);
    }

    // --- decoder --------------------------------------------------------
    const size_t S0_B = (size_t)34 * 34 * 256 * 128 * 2;
    hipMemsetAsync(A0, 0, S0_B, stream);
    conv(ZQ, F_qp, postq_b, A0, nullptr, 16, 16, 256, 16, 16, 256, 1, 1,
         34, (long)34 * 34 * 256, 1, 1, 2, 0, 0);        // postq -> zero-stuffed
    hipMemsetAsync(A1, 0, A1_B, stream);
    conv(A0, F_d1, dec_b1, A1, nullptr, 34, 34, 256, 32, 32, 64, 1, 3,
         66, (long)66 * 66 * 64, 1, 1, 2, 1, 0);         // convT: 16->32, stuffed out
    conv(A1, F_d2, dec_b2, A0, nullptr, 66, 66, 64, 64, 64, 64, 1, 3,
         64, (long)64 * 64 * 64, 0, 0, 1, 1, 0);         // convT: 32->64
    conv(A0, F_d3, dec_b3, A1, nullptr, 64, 64, 64, 62, 62, 64, 1, 3,
         62, (long)62 * 62 * 64, 0, 0, 1, 1, 0);         // 64->62 (pad 0)
    conv(A1, F_ow, out_b, nullptr, logits, 62, 62, 64, 62, 62, 68, 1, 1,
         62, 0, 0, 0, 1, 0, 1);                          // logits f32 NHWC
}
